// SparseMultiHopMoE_5205500362915
// MI455X (gfx1250) — matrix operand order repaired and checked by reading, not compiled, not run on hardware
//
#include <hip/hip_runtime.h>
#include <hip/hip_bf16.h>

// ---------------------------------------------------------------------------
// SparseMultiHopMoE for MI455X (gfx1250, wave32, WMMA f16 16x16x32).
// A^2 / A^3 dense 4096^3 GEMMs dominate -> v_wmma_f32_16x16x32_f16 with
// double-buffered GLOBAL_LOAD_ASYNC_TO_LDS_B128 staging (ASYNCcnt path).
// Counts are small integers (exact in f16 inputs, f32 accumulation).
// softmax(log1p(cn)) == (1+cn)/sum(1+cn) over kept entries (exp-free).
// ---------------------------------------------------------------------------

#define NNODES 4096
#define DINF   64
#define DHF    128
#define TOPKK  50

#define BKS        64          // K-slab per LDS stage
#define LDSROW     80          // halves per LDS row (64 data + 16 pad, 160B)
#define TILE_HALF  (128 * LDSROW)

typedef __attribute__((ext_vector_type(16))) _Float16 v16h;
typedef __attribute__((ext_vector_type(8)))  _Float16 v8h;
typedef __attribute__((ext_vector_type(8)))  float    v8f;

// Low 32 bits of a generic pointer to LDS == AS(3) byte offset.
__device__ __forceinline__ unsigned lds_off(const void* p) {
  return (unsigned)(unsigned long long)(uintptr_t)p;
}

__device__ __forceinline__ void async_b128(unsigned ldsByteOff, const void* gaddr) {
  asm volatile("global_load_async_to_lds_b128 %0, %1, off"
               :: "v"(ldsByteOff), "v"(gaddr) : "memory");
}
__device__ __forceinline__ void wait_async0() {
  asm volatile("s_wait_asynccnt 0x0" ::: "memory");
}

// ---------------------------------------------------------------------------
// Power GEMM (f16 out): C[M x Nout](f16) = A[M x K](f16,row-major) x B where
// B is supplied as BT[n*K + k] (row-major B^T). LDS double-buffered async.
// Block: 256 threads = 8 waves (4M x 2N), wave tile 32x64, block tile 128x128.
// ---------------------------------------------------------------------------
__global__ __launch_bounds__(256) void gemm_pow_kernel(
    const _Float16* __restrict__ A, const _Float16* __restrict__ BT,
    _Float16* __restrict__ C, int M, int Nout, int K)
{
  __shared__ _Float16 smem[2][2][TILE_HALF];   // [buf][A/B][128*LDSROW]

  const int tid  = threadIdx.x;
  const int lane = tid & 31;
  const int wave = tid >> 5;
  const int rowW = (wave & 3) * 32;            // wave M offset in block tile
  const int colW = (wave >> 2) * 64;           // wave N offset in block tile
  const int row0 = blockIdx.y * 128;
  const int col0 = blockIdx.x * 128;

  const unsigned sbase     = lds_off(&smem[0][0][0]);
  const unsigned tileBytes = TILE_HALF * 2;

  v8f acc[2][4];
#pragma unroll
  for (int a = 0; a < 2; a++)
#pragma unroll
    for (int b = 0; b < 4; b++)
#pragma unroll
      for (int r = 0; r < 8; r++) acc[a][b][r] = 0.0f;

  // A-frag (16x32): lanes 0-15 -> M=lane, K 0..15; 16-31 -> M-16, K 16..31
  const int aM  = lane & 15;
  const int aKo = (lane >> 4) << 4;            // 0 or 16
  // B-frag (32x16) from BT rows: lanes 0-15 -> col n=lane, K 0..15; 16-31 -> n-16, K 16..31
  const int bN  = lane & 15;
  const int bKo = (lane >> 4) << 4;            // 0 or 16

  // issue one K-slab: A rows [row0,+128) x [k0,k0+64), BT rows [col0,+128)
  auto issue = [&](int buf, int k0) {
#pragma unroll
    for (int c = 0; c < 4; c++) {
      const int chunk = tid + c * 256;         // 1024 chunks of 16B per tile
      const int r   = chunk >> 3;
      const int seg = chunk & 7;
      const unsigned la = sbase + (unsigned)buf * 2u * tileBytes +
                          (unsigned)r * (LDSROW * 2) + (unsigned)seg * 16;
      const void* ga = (const char*)A + ((size_t)(row0 + r) * (size_t)K + k0 + seg * 8) * 2;
      async_b128(la, ga);
      const unsigned lb = la + tileBytes;
      const void* gb = (const char*)BT + ((size_t)(col0 + r) * (size_t)K + k0 + seg * 8) * 2;
      async_b128(lb, gb);
    }
  };

  issue(0, 0);
  wait_async0();
  __syncthreads();

  const int nk = K / BKS;
  for (int kt = 0; kt < nk; ++kt) {
    const int buf = kt & 1;
    if (kt + 1 < nk) issue(buf ^ 1, (kt + 1) * BKS);

    const _Float16* As = &smem[buf][0][0];
    const _Float16* Bs = &smem[buf][1][0];
#pragma unroll
    for (int ks = 0; ks < BKS; ks += 32) {
      v16h afrag[2];
#pragma unroll
      for (int mi = 0; mi < 2; mi++) {
        const _Float16* p = As + (rowW + mi * 16 + aM) * LDSROW + ks + aKo;
        v8h lo = *(const v8h*)(p);
        v8h hi = *(const v8h*)(p + 8);
        v16h t;
#pragma unroll
        for (int e = 0; e < 8; e++) { t[e] = lo[e]; t[e + 8] = hi[e]; }
        afrag[mi] = t;
      }
      v16h bfrag[4];
#pragma unroll
      for (int ni = 0; ni < 4; ni++) {
        const _Float16* p = Bs + (colW + ni * 16 + bN) * LDSROW + ks + bKo;
        v8h lo = *(const v8h*)(p);
        v8h hi = *(const v8h*)(p + 8);
        v16h t;
#pragma unroll
        for (int e = 0; e < 8; e++) { t[e] = lo[e]; t[e + 8] = hi[e]; }
        bfrag[ni] = t;
      }
#pragma unroll
      for (int mi = 0; mi < 2; mi++)
#pragma unroll
        for (int ni = 0; ni < 4; ni++)
          acc[mi][ni] = __builtin_amdgcn_wmma_f32_16x16x32_f16(
              false, afrag[mi], false, bfrag[ni], (short)0, acc[mi][ni], false, false);
    }
    wait_async0();        // own slab for next buffer landed
    __syncthreads();      // whole workgroup done reading current buffer
  }

  // C-frag (16x16 f32): VGPR r: lanes 0-15 -> (M=r,N=lane); 16-31 -> (M=8+r,N=lane-16)
  const int cN  = lane & 15;
  const int cMo = (lane >> 4) << 3;
#pragma unroll
  for (int mi = 0; mi < 2; mi++)
#pragma unroll
    for (int ni = 0; ni < 4; ni++) {
      const int n     = col0 + colW + ni * 16 + cN;
      const int mbase = row0 + rowW + mi * 16 + cMo;
#pragma unroll
      for (int r = 0; r < 8; r++)
        C[(size_t)(mbase + r) * (size_t)Nout + n] = (_Float16)acc[mi][ni][r];
    }
}

// ---------------------------------------------------------------------------
// Direct-global WMMA GEMM for the skinny (Nout=128) GEMMs.
// MODE 1: C(f32) += result        (MoE accumulation across hops)
// MODE 2: C(f32) = relu(result + bias[n])   (GCN branch)
// ---------------------------------------------------------------------------
template <int MODE>
__global__ __launch_bounds__(256) void gemm_wmma_kernel(
    const _Float16* __restrict__ A, const _Float16* __restrict__ BT,
    float* __restrict__ C, const float* __restrict__ bias,
    int M, int Nout, int K)
{
  const int lane = threadIdx.x & 31;
  const int wave = threadIdx.x >> 5;
  const int row0 = blockIdx.y * 128 + (wave & 3) * 32;
  const int col0 = blockIdx.x * 128 + (wave >> 2) * 64;

  v8f acc[2][4];
#pragma unroll
  for (int a = 0; a < 2; a++)
#pragma unroll
    for (int b = 0; b < 4; b++)
#pragma unroll
      for (int r = 0; r < 8; r++) acc[a][b][r] = 0.0f;

  const int aM  = lane & 15;
  const int aKo = (lane >> 4) << 4;
  const int bN  = lane & 15;
  const int bKo = (lane >> 4) << 4;

  for (int k0 = 0; k0 < K; k0 += 32) {
    v16h afrag[2];
#pragma unroll
    for (int mi = 0; mi < 2; mi++) {
      const _Float16* p = A + (size_t)(row0 + mi * 16 + aM) * (size_t)K + (k0 + aKo);
      v8h lo = *(const v8h*)(p);
      v8h hi = *(const v8h*)(p + 8);
      v16h t;
#pragma unroll
      for (int e = 0; e < 8; e++) { t[e] = lo[e]; t[e + 8] = hi[e]; }
      afrag[mi] = t;
    }
    v16h bfrag[4];
#pragma unroll
    for (int ni = 0; ni < 4; ni++)
      bfrag[ni] = *(const v16h*)(BT + (size_t)(col0 + ni * 16 + bN) * (size_t)K + (k0 + bKo));

#pragma unroll
    for (int mi = 0; mi < 2; mi++)
#pragma unroll
      for (int ni = 0; ni < 4; ni++)
        acc[mi][ni] = __builtin_amdgcn_wmma_f32_16x16x32_f16(
            false, afrag[mi], false, bfrag[ni], (short)0, acc[mi][ni], false, false);
  }

  const int cN  = lane & 15;
  const int cMo = (lane >> 4) << 3;
#pragma unroll
  for (int mi = 0; mi < 2; mi++) {
#pragma unroll
    for (int ni = 0; ni < 4; ni++) {
      const int n     = col0 + ni * 16 + cN;
      const int mbase = row0 + mi * 16 + cMo;
      if constexpr (MODE == 1) {
#pragma unroll
        for (int r = 0; r < 8; r++) {
          size_t idx = (size_t)(mbase + r) * (size_t)Nout + n;
          C[idx] = C[idx] + acc[mi][ni][r];
        }
      } else {
        const float bb = bias ? bias[n] : 0.0f;
#pragma unroll
        for (int r = 0; r < 8; r++) {
          size_t idx = (size_t)(mbase + r) * (size_t)Nout + n;
          float v = acc[mi][ni][r] + bb;
          C[idx] = v > 0.0f ? v : 0.0f;
        }
      }
    }
  }
}

// ---------------------------------------------------------------------------
// adj (f32) -> adjH (f16 row-major) + adjT (f16 transpose)
// ---------------------------------------------------------------------------
__global__ __launch_bounds__(256) void cvt_adj_kernel(
    const float* __restrict__ adj, _Float16* __restrict__ adjH,
    _Float16* __restrict__ adjT)
{
  size_t idx = (size_t)blockIdx.x * 256 + threadIdx.x;
  int i = (int)(idx >> 12);
  int j = (int)(idx & 4095);
  float a = adj[idx];
  adjH[idx] = (_Float16)a;
  adjT[(size_t)j * NNODES + i] = (_Float16)a;
}

__global__ __launch_bounds__(256) void deg_kernel(
    const float* __restrict__ adj, float* __restrict__ dinv)
{
  __shared__ float s;
  const int row = blockIdx.x;
  if (threadIdx.x == 0) s = 0.0f;
  __syncthreads();
  float l = 0.0f;
  for (int j = threadIdx.x; j < NNODES; j += 256)
    l += adj[(size_t)row * NNODES + j];
  atomicAdd(&s, l);
  __syncthreads();
  if (threadIdx.x == 0) dinv[row] = rsqrtf(fmaxf(s, 1.0f));
}

__global__ __launch_bounds__(256) void anorm_kernel(
    const float* __restrict__ adj, const float* __restrict__ dinv,
    _Float16* __restrict__ an)
{
  size_t idx = (size_t)blockIdx.x * 256 + threadIdx.x;
  int i = (int)(idx >> 12);
  int j = (int)(idx & 4095);
  an[idx] = (_Float16)(adj[idx] * dinv[i] * dinv[j]);
}

// outT[j*N + i] = sum_k x[i,k] * W[k,j]   (transposed f16 output, B^T layout)
__global__ __launch_bounds__(256) void xw_kernel(
    const float* __restrict__ x, const float* __restrict__ w,
    _Float16* __restrict__ outT)
{
  int idx = blockIdx.x * 256 + threadIdx.x;
  int i = idx >> 7;
  int j = idx & (DHF - 1);
  float s = 0.0f;
#pragma unroll 8
  for (int k = 0; k < DINF; k++) s += x[(size_t)i * DINF + k] * w[(size_t)k * DHF + j];
  outT[(size_t)j * NNODES + i] = (_Float16)s;
}

__global__ __launch_bounds__(256) void zero_f32_kernel(float* __restrict__ p) {
  p[(size_t)blockIdx.x * 256 + threadIdx.x] = 0.0f;
}

// ---------------------------------------------------------------------------
// Per-row CN attention (counts exact in f16). top-k(50) via integer-threshold
// binary search; softmax(log1p) == (1+cn)/S. One 256-thread block per row.
// ---------------------------------------------------------------------------
__global__ __launch_bounds__(256) void attn_kernel(
    const _Float16* __restrict__ CN, _Float16* __restrict__ OUT, int do_topk)
{
  __shared__ int   s_cnt;
  __shared__ float s_sum;
  const int row = blockIdx.x;
  const int tid = threadIdx.x;

  float v[16];
#pragma unroll
  for (int e = 0; e < 16; e++) {
    int j = tid + e * 256;
    float c = (float)CN[(size_t)row * NNODES + j];
    if (j == row) c = 0.0f;
    v[e] = c;
  }

  float thr = 1.0f;
  if (do_topk) {
    int lo = 1, hi = NNODES, t = 1;
    while (lo <= hi) {
      int mid = (lo + hi) >> 1;
      if (tid == 0) s_cnt = 0;
      __syncthreads();
      int c = 0;
#pragma unroll
      for (int e = 0; e < 16; e++) c += (v[e] >= (float)mid) ? 1 : 0;
      atomicAdd(&s_cnt, c);
      __syncthreads();
      int total = s_cnt;
      __syncthreads();
      if (total >= TOPKK) { t = mid; lo = mid + 1; } else { hi = mid - 1; }
    }
    thr = (float)t;
  }

  if (tid == 0) s_sum = 0.0f;
  __syncthreads();
  float ls = 0.0f;
#pragma unroll
  for (int e = 0; e < 16; e++)
    if (v[e] >= thr) ls += 1.0f + v[e];
  atomicAdd(&s_sum, ls);
  __syncthreads();
  const float S   = s_sum;
  const float inv = (S > 0.0f) ? (1.0f / S) : 0.0f;
#pragma unroll
  for (int e = 0; e < 16; e++) {
    int j = tid + e * 256;
    float a = (v[e] >= thr) ? (1.0f + v[e]) * inv : 0.0f;
    OUT[(size_t)row * NNODES + j] = (_Float16)a;
  }
}

// ---------------------------------------------------------------------------
// Gated fusion. One 128-thread block per node.
// ---------------------------------------------------------------------------
__global__ __launch_bounds__(128) void fusion_kernel(
    const float* __restrict__ gcn, const float* __restrict__ moeAcc,
    const float* __restrict__ g1w, const float* __restrict__ g1b,
    const float* __restrict__ g2w, const float* __restrict__ g2b,
    float* __restrict__ out)
{
  __shared__ float gi[2 * DHF];
  __shared__ float red[DHF];
  const int i = blockIdx.x;
  const int t = threadIdx.x;
  const float g = gcn[(size_t)i * DHF + t];
  const float m = moeAcc[(size_t)i * DHF + t] * (1.0f / 3.0f);
  gi[t] = g;
  gi[DHF + t] = m;
  __syncthreads();
  float h = g1b[t];
  for (int c = 0; c < 2 * DHF; c++) h += gi[c] * g1w[(size_t)c * DHF + t];
  h = fmaxf(h, 0.0f);
  red[t] = h * g2w[t];
  __syncthreads();
  for (int s = DHF / 2; s > 0; s >>= 1) {
    if (t < s) red[t] += red[t + s];
    __syncthreads();
  }
  const float logit = red[0] + g2b[0];
  const float ratio = 1.0f / (1.0f + expf(-logit));
  const float w     = 0.1f + 0.9f * ratio;
  out[(size_t)i * DHF + t] = w * g + (1.0f - w) * m;
}

// ---------------------------------------------------------------------------
extern "C" void kernel_launch(void* const* d_in, const int* in_sizes, int n_in,
                              void* d_out, int out_size, void* d_ws, size_t ws_size,
                              hipStream_t stream) {
  const float* x     = (const float*)d_in[0];
  const float* adj   = (const float*)d_in[1];
  const float* w_gcn = (const float*)d_in[2];
  const float* b_gcn = (const float*)d_in[3];
  const float* w_exp = (const float*)d_in[4];
  const float* g1_w  = (const float*)d_in[5];
  const float* g1_b  = (const float*)d_in[6];
  const float* g2_w  = (const float*)d_in[7];
  const float* g2_b  = (const float*)d_in[8];
  float* out = (float*)d_out;

  const size_t NN2 = (size_t)NNODES * NNODES;
  _Float16* adjH = (_Float16*)d_ws;
  _Float16* adjT = adjH + NN2;
  _Float16* bufA = adjT + NN2;
  _Float16* bufB = bufA + NN2;
  _Float16* xeT0 = bufB + NN2;
  _Float16* xeT1 = xeT0 + (size_t)DHF * NNODES;
  _Float16* xeT2 = xeT1 + (size_t)DHF * NNODES;
  _Float16* xgT  = xeT2 + (size_t)DHF * NNODES;
  float*    dinv   = (float*)(xgT + (size_t)DHF * NNODES);
  float*    moeAcc = dinv + NNODES;
  float*    gcnOut = moeAcc + (size_t)NNODES * DHF;

  const dim3 blk256(256), blk128(128);
  const int  elemBlocks = (int)(NN2 / 256);
  const dim3 gBig(NNODES / 128, NNODES / 128);
  const dim3 gMoe(DHF / 128, NNODES / 128);

  // ---- preparation ----
  cvt_adj_kernel<<<elemBlocks, blk256, 0, stream>>>(adj, adjH, adjT);
  deg_kernel<<<NNODES, blk256, 0, stream>>>(adj, dinv);
  xw_kernel<<<(NNODES * DHF) / 256, blk256, 0, stream>>>(x, w_exp + 0 * DINF * DHF, xeT0);
  xw_kernel<<<(NNODES * DHF) / 256, blk256, 0, stream>>>(x, w_exp + 1 * DINF * DHF, xeT1);
  xw_kernel<<<(NNODES * DHF) / 256, blk256, 0, stream>>>(x, w_exp + 2 * DINF * DHF, xeT2);
  xw_kernel<<<(NNODES * DHF) / 256, blk256, 0, stream>>>(x, w_gcn, xgT);
  zero_f32_kernel<<<(NNODES * DHF) / 256, blk256, 0, stream>>>(moeAcc);

  // ---- hop 1 ----
  attn_kernel<<<NNODES, blk256, 0, stream>>>(adjH, bufB, 0);
  gemm_wmma_kernel<1><<<gMoe, blk256, 0, stream>>>(bufB, xeT0, moeAcc, nullptr,
                                                   NNODES, DHF, NNODES);
  // ---- A^2 = A @ A (async-LDS WMMA) ----
  gemm_pow_kernel<<<gBig, blk256, 0, stream>>>(adjH, adjT, bufA, NNODES, NNODES, NNODES);
  // ---- hop 2 ----
  attn_kernel<<<NNODES, blk256, 0, stream>>>(bufA, bufB, 1);
  gemm_wmma_kernel<1><<<gMoe, blk256, 0, stream>>>(bufB, xeT1, moeAcc, nullptr,
                                                   NNODES, DHF, NNODES);
  // ---- A^3 = A^2 @ A ----
  gemm_pow_kernel<<<gBig, blk256, 0, stream>>>(bufA, adjT, bufB, NNODES, NNODES, NNODES);
  // ---- hop 3 ----
  attn_kernel<<<NNODES, blk256, 0, stream>>>(bufB, bufA, 1);
  gemm_wmma_kernel<1><<<gMoe, blk256, 0, stream>>>(bufA, xeT2, moeAcc, nullptr,
                                                   NNODES, DHF, NNODES);

  // ---- GCN branch ----
  anorm_kernel<<<elemBlocks, blk256, 0, stream>>>(adj, dinv, bufB);
  gemm_wmma_kernel<2><<<gMoe, blk256, 0, stream>>>(bufB, xgT, gcnOut, b_gcn,
                                                   NNODES, DHF, NNODES);

  // ---- gated fusion ----
  fusion_kernel<<<NNODES, blk128, 0, stream>>>(gcnOut, moeAcc, g1_w, g1_b,
                                               g2_w, g2_b, out);
}
